// PointNetSetAbstraction_24111946399909
// MI455X (gfx1250) — matrix operand order repaired and checked by reading, not compiled, not run on hardware
//
#include <hip/hip_runtime.h>
#include <hip/hip_bf16.h>

typedef _Float16 fp16_t;
typedef __attribute__((ext_vector_type(16))) _Float16 v16h;
typedef __attribute__((ext_vector_type(8)))  float    v8f;

// Problem constants (match reference)
constexpr int B_  = 16;
constexpr int N_  = 4096;
constexpr int S_  = 1024;        // NPOINT
constexpr int K_  = 32;          // NSAMPLE
constexpr long long M_ = (long long)B_ * S_ * K_;   // 524288 positions
constexpr float RADIUS2 = 0.2f * 0.2f;
constexpr float BIGF = 1e10f;
constexpr float EPSF = 1e-5f;
constexpr int HALFW = 64;        // DIMSORT_RANGE/2

// ---------------------------------------------------------------------------
// 1) Windowed FPS. One block per batch; dist[] lives in LDS; a 32-entry
//    segment-max tree makes each sequential step O(256) instead of O(4096).
// ---------------------------------------------------------------------------
__global__ __launch_bounds__(256) void fps_kernel(const float* __restrict__ xyz,
                                                  const int* __restrict__ start,
                                                  int* __restrict__ fps_idx) {
  __shared__ float dist[N_];        // 16 KB
  __shared__ float rv[256];
  __shared__ int   ri[256];
  __shared__ float segV[32];
  __shared__ int   segI[32];
  __shared__ int   sFar;

  const int b   = blockIdx.x;
  const int tid = threadIdx.x;
  const float* xb = xyz + (size_t)b * 3 * N_;

  for (int i = tid; i < N_; i += 256) dist[i] = BIGF;
  if (tid < 32) { segV[tid] = BIGF; segI[tid] = tid * 128; }
  if (tid == 0) sFar = start[b];
  __syncthreads();

  for (int step = 0; step < S_; ++step) {
    const int far = sFar;
    if (tid == 0) fps_idx[b * S_ + step] = far;
    const float cx = xb[far], cy = xb[N_ + far], cz = xb[2 * N_ + far];
    const int upper = (far <= N_ - HALFW) ? far + HALFW : N_;
    const int lower = (far > HALFW) ? far - HALFW : 0;

    // window update (<=128 entries; dist only decreases)
    const int wi = lower + tid;
    if (tid < 128 && wi < upper) {
      const float dx = xb[wi] - cx, dy = xb[N_ + wi] - cy, dz = xb[2 * N_ + wi] - cz;
      dist[wi] = fminf(dist[wi], dx * dx + dy * dy + dz * dz);
    }
    __syncthreads();

    // recompute the (<=2) 128-wide segments the window touched
    const int seg0 = lower >> 7;
    const int seg1 = (upper - 1) >> 7;
    {
      const int sg = (tid < 128) ? seg0 : seg1;
      const int j  = tid & 127;
      const int idx = sg * 128 + j;
      rv[tid] = dist[idx];
      ri[tid] = idx;
    }
    __syncthreads();
    #pragma unroll
    for (int off = 64; off > 0; off >>= 1) {
      if ((tid & 127) < off) {
        const float v2 = rv[tid + off]; const int i2 = ri[tid + off];
        if (v2 > rv[tid] || (v2 == rv[tid] && i2 < ri[tid])) { rv[tid] = v2; ri[tid] = i2; }
      }
      __syncthreads();
    }
    if (tid == 0)   { segV[seg0] = rv[0];   segI[seg0] = ri[0]; }
    if (tid == 128) { segV[seg1] = rv[128]; segI[seg1] = ri[128]; }
    __syncthreads();

    if (tid == 0) {
      float bv = -1.0f; int bi = 0;
      #pragma unroll
      for (int sg = 0; sg < 32; ++sg) {
        if (segV[sg] > bv) { bv = segV[sg]; bi = segI[sg]; }
      }
      sFar = bi;
    }
    __syncthreads();
  }
}

// ---------------------------------------------------------------------------
// 2) Ball query + grouping. One wave32 per (b,s); sequential 32-wide scan with
//    ballot keeps the first-32-in-index-order semantics of the reference.
//    Writes f16 activations position-major [M][32] (ch 0-2 gnorm, 3-5 points,
//    6-31 zero-pad so layer-1 K fits one WMMA K=32 step). Also emits new_xyz.
// ---------------------------------------------------------------------------
__global__ __launch_bounds__(256) void group_kernel(const float* __restrict__ xyz,
                                                    const float* __restrict__ pts,
                                                    const int* __restrict__ fps_idx,
                                                    fp16_t* __restrict__ X0,
                                                    float* __restrict__ new_xyz) {
  const int lane = threadIdx.x & 31;
  const int wave = blockIdx.x * (blockDim.x >> 5) + (threadIdx.x >> 5);
  if (wave >= B_ * S_) return;
  const int b = wave >> 10;
  const int s = wave & (S_ - 1);
  const float* xb = xyz + (size_t)b * 3 * N_;
  const float* pb = pts + (size_t)b * 3 * N_;
  const int ci = fps_idx[b * S_ + s];
  const float cx = xb[ci], cy = xb[N_ + ci], cz = xb[2 * N_ + ci];
  if (lane < 3) {
    const float v = (lane == 0) ? cx : ((lane == 1) ? cy : cz);
    new_xyz[((size_t)b * 3 + lane) * S_ + s] = v;
  }

  int cnt = 0;
  int firstIdx = -1;
  for (int base = 0; base < N_ && cnt < K_; base += 32) {
    const int idx = base + lane;
    const float dx = xb[idx] - cx, dy = xb[N_ + idx] - cy, dz = xb[2 * N_ + idx] - cz;
    const bool hit = (dx * dx + dy * dy + dz * dz) <= RADIUS2;
    const unsigned bal = (unsigned)__ballot(hit);
    if (firstIdx < 0 && bal) firstIdx = base + (__ffs(bal) - 1);
    const int rank = __popc(bal & ((1u << lane) - 1u));
    if (hit) {
      const int slot = cnt + rank;
      if (slot < K_) {
        const size_t m = ((size_t)(b * S_ + s)) * K_ + slot;
        fp16_t* dst = X0 + m * 32;
        dst[0] = (fp16_t)dx; dst[1] = (fp16_t)dy; dst[2] = (fp16_t)dz;
        dst[3] = (fp16_t)pb[idx]; dst[4] = (fp16_t)pb[N_ + idx]; dst[5] = (fp16_t)pb[2 * N_ + idx];
        #pragma unroll
        for (int c = 6; c < 32; ++c) dst[c] = (fp16_t)0.0f;
      }
    }
    cnt += __popc(bal);
  }
  if (cnt < K_) {                       // fill misses with first hit
    if (firstIdx < 0) firstIdx = ci;    // centroid always within radius
    if (lane >= cnt && lane < K_) {
      const int idx = firstIdx;
      const float dx = xb[idx] - cx, dy = xb[N_ + idx] - cy, dz = xb[2 * N_ + idx] - cz;
      const size_t m = ((size_t)(b * S_ + s)) * K_ + lane;
      fp16_t* dst = X0 + m * 32;
      dst[0] = (fp16_t)dx; dst[1] = (fp16_t)dy; dst[2] = (fp16_t)dz;
      dst[3] = (fp16_t)pb[idx]; dst[4] = (fp16_t)pb[N_ + idx]; dst[5] = (fp16_t)pb[2 * N_ + idx];
      #pragma unroll
      for (int c = 6; c < 32; ++c) dst[c] = (fp16_t)0.0f;
    }
  }
}

// ---------------------------------------------------------------------------
// 3) WMMA GEMM: raw[Cout][M] = W[Cout,Cin] * act[M][CINPAD]^T + bias.
//    One wave handles 16 positions x all Cout rows; A (weights) preloaded into
//    fragments; B is one contiguous v16h load per K-step. f16 in, f32 accum.
// ---------------------------------------------------------------------------
template <int CINPAD, int COUT>
__global__ __launch_bounds__(256) void gemm_kernel(const fp16_t* __restrict__ act,
                                                   const float* __restrict__ W,
                                                   const float* __restrict__ bias,
                                                   float* __restrict__ raw,
                                                   int Cin) {
  constexpr int KSTEPS = CINPAD / 32;
  constexpr int ROWT   = COUT / 16;
  const int lane = threadIdx.x & 31;
  const int w    = threadIdx.x >> 5;
  const int pbase = blockIdx.x * 128 + w * 16;
  const int r  = lane & 15;
  const int hi = lane >> 4;
  const int pos = pbase + r;

  // A fragments: contiguous k order, matching B (lanes 0-15: K 0-15; 16-31: K 16-31)
  v16h A[ROWT][KSTEPS];
  #pragma unroll
  for (int rt = 0; rt < ROWT; ++rt) {
    const int row = rt * 16 + r;
    #pragma unroll
    for (int ks = 0; ks < KSTEPS; ++ks) {
      v16h a;
      #pragma unroll
      for (int v = 0; v < 8; ++v) {
        const int kk = 2 * v + (hi ? 16 : 0) + ks * 32;
        a[2 * v]     = (kk     < Cin) ? (fp16_t)W[row * Cin + kk]     : (fp16_t)0.0f;
        a[2 * v + 1] = (kk + 1 < Cin) ? (fp16_t)W[row * Cin + kk + 1] : (fp16_t)0.0f;
      }
      A[rt][ks] = a;
    }
  }

  v8f acc[ROWT] = {};
  const fp16_t* actp = act + (size_t)pos * CINPAD + (hi ? 16 : 0);
  __builtin_prefetch(act + (size_t)(pos + 128) * CINPAD, 0, 1);  // global_prefetch_b8

  #pragma unroll
  for (int ks = 0; ks < KSTEPS; ++ks) {
    const v16h bfrag = *(const v16h*)(actp + ks * 32);   // contiguous 32B per lane
    #pragma unroll
    for (int rt = 0; rt < ROWT; ++rt) {
      acc[rt] = __builtin_amdgcn_wmma_f32_16x16x32_f16(
          false, A[rt][ks], false, bfrag, (short)0, acc[rt], false, false);
    }
  }

  // Epilogue: D layout — VGPR rr holds rows rr (lanes 0-15) / rr+8 (lanes 16-31)
  #pragma unroll
  for (int rt = 0; rt < ROWT; ++rt) {
    #pragma unroll
    for (int rr = 0; rr < 8; ++rr) {
      const int ch = rt * 16 + (hi ? 8 : 0) + rr;
      raw[(size_t)ch * M_ + pos] = acc[rt][rr] + bias[ch];
    }
  }
}

// ---------------------------------------------------------------------------
// 4) Per-channel sum / sumsq (one block per channel, coalesced stream of M).
// ---------------------------------------------------------------------------
__global__ __launch_bounds__(256) void stats_kernel(const float* __restrict__ raw,
                                                    float* __restrict__ sum,
                                                    float* __restrict__ sumsq) {
  __shared__ float s1[256], s2[256];
  const int c = blockIdx.x, tid = threadIdx.x;
  const float* p = raw + (size_t)c * M_;
  float a = 0.f, q = 0.f;
  for (long long m = tid; m < M_; m += 256) { const float v = p[m]; a += v; q += v * v; }
  s1[tid] = a; s2[tid] = q; __syncthreads();
  for (int off = 128; off > 0; off >>= 1) {
    if (tid < off) { s1[tid] += s1[tid + off]; s2[tid] += s2[tid + off]; }
    __syncthreads();
  }
  if (tid == 0) { sum[c] = s1[0]; sumsq[c] = s2[0]; }
}

__global__ void bnparams_kernel(const float* __restrict__ sum, const float* __restrict__ sumsq,
                                const float* __restrict__ g, const float* __restrict__ be,
                                float* __restrict__ scale, float* __restrict__ shift, int Cout) {
  const int c = threadIdx.x;
  if (c < Cout) {
    const float inv = 1.0f / (float)M_;
    const float mean = sum[c] * inv;
    const float var  = sumsq[c] * inv - mean * mean;
    const float a = g[c] * rsqrtf(var + EPSF);
    scale[c] = a;
    shift[c] = be[c] - mean * a;
  }
}

// BN + ReLU, transposing channel-major f32 -> position-major f16 for next GEMM
template <int COUT>
__global__ __launch_bounds__(256) void bnrelu_kernel(const float* __restrict__ raw,
                                                     const float* __restrict__ scale,
                                                     const float* __restrict__ shift,
                                                     fp16_t* __restrict__ actout) {
  __shared__ float sS[COUT], sH[COUT];
  for (int c = threadIdx.x; c < COUT; c += 256) { sS[c] = scale[c]; sH[c] = shift[c]; }
  __syncthreads();
  const long long m = (long long)blockIdx.x * 256 + threadIdx.x;
  if (m < M_) {
    fp16_t* dst = actout + (size_t)m * COUT;
    #pragma unroll 8
    for (int c = 0; c < COUT; ++c) {
      const float v = raw[(size_t)c * M_ + m] * sS[c] + sH[c];
      dst[c] = (fp16_t)fmaxf(v, 0.0f);
    }
  }
}

// ---------------------------------------------------------------------------
// 5) Fused layer-3 BN + ReLU + max over K -> feat [B,128,S]
// ---------------------------------------------------------------------------
__global__ __launch_bounds__(256) void maxpool_kernel(const float* __restrict__ raw,
                                                      const float* __restrict__ scale,
                                                      const float* __restrict__ shift,
                                                      float* __restrict__ feat) {
  const int idx = blockIdx.x * 256 + threadIdx.x;       // [128][16][1024]
  if (idx >= 128 * B_ * S_) return;
  const int s = idx & (S_ - 1);
  const int rem = idx >> 10;
  const int b = rem & (B_ - 1);
  const int c = rem >> 4;
  const float a = scale[c], h = shift[c];
  const float* p = raw + (size_t)c * M_ + ((size_t)(b * S_ + s)) * K_;
  float mx = -BIGF;
  #pragma unroll
  for (int k = 0; k < K_; ++k) mx = fmaxf(mx, p[k] * a + h);
  mx = fmaxf(mx, 0.0f);                                 // relu is monotone
  feat[((size_t)b * 128 + c) * S_ + s] = mx;
}

// ---------------------------------------------------------------------------
extern "C" void kernel_launch(void* const* d_in, const int* in_sizes, int n_in,
                              void* d_out, int out_size, void* d_ws, size_t ws_size,
                              hipStream_t stream) {
  (void)in_sizes; (void)n_in; (void)out_size; (void)ws_size;
  const float* xyz = (const float*)d_in[0];
  const float* pts = (const float*)d_in[1];
  const int* start = (const int*)d_in[2];
  const float* w0 = (const float*)d_in[3];  const float* b0 = (const float*)d_in[4];
  const float* g0 = (const float*)d_in[5];  const float* be0 = (const float*)d_in[6];
  const float* w1 = (const float*)d_in[7];  const float* b1 = (const float*)d_in[8];
  const float* g1 = (const float*)d_in[9];  const float* be1 = (const float*)d_in[10];
  const float* w2 = (const float*)d_in[11]; const float* b2 = (const float*)d_in[12];
  const float* g2 = (const float*)d_in[13]; const float* be2 = (const float*)d_in[14];

  char* ws = (char*)d_ws;
  size_t off = 0;
  auto alloc = [&](size_t bytes) -> void* {
    void* p = ws + off;
    off += (bytes + 255) & ~(size_t)255;
    return p;
  };
  int*    fps_idx = (int*)alloc((size_t)B_ * S_ * sizeof(int));
  fp16_t* X0  = (fp16_t*)alloc((size_t)M_ * 32 * sizeof(fp16_t));
  fp16_t* Y1  = (fp16_t*)alloc((size_t)M_ * 64 * sizeof(fp16_t));
  fp16_t* Y2  = (fp16_t*)alloc((size_t)M_ * 64 * sizeof(fp16_t));
  float*  raw = (float*)alloc((size_t)M_ * 128 * sizeof(float));  // reused per layer
  float*  ssum  = (float*)alloc(128 * sizeof(float));
  float*  ssq   = (float*)alloc(128 * sizeof(float));
  float*  scale = (float*)alloc(128 * sizeof(float));
  float*  shift = (float*)alloc(128 * sizeof(float));

  float* new_xyz = (float*)d_out;                  // [B,3,S]
  float* feat    = (float*)d_out + (size_t)B_ * 3 * S_;  // [B,128,S]

  // FPS + grouping
  fps_kernel<<<B_, 256, 0, stream>>>(xyz, start, fps_idx);
  group_kernel<<<(B_ * S_) / 8, 256, 0, stream>>>(xyz, pts, fps_idx, X0, new_xyz);

  const int gemm_blocks = (int)(M_ / 128);
  const int bn_blocks   = (int)(M_ / 256);

  // Layer 1: 6 (pad 32) -> 64
  gemm_kernel<32, 64><<<gemm_blocks, 256, 0, stream>>>(X0, w0, b0, raw, 6);
  stats_kernel<<<64, 256, 0, stream>>>(raw, ssum, ssq);
  bnparams_kernel<<<1, 128, 0, stream>>>(ssum, ssq, g0, be0, scale, shift, 64);
  bnrelu_kernel<64><<<bn_blocks, 256, 0, stream>>>(raw, scale, shift, Y1);

  // Layer 2: 64 -> 64
  gemm_kernel<64, 64><<<gemm_blocks, 256, 0, stream>>>(Y1, w1, b1, raw, 64);
  stats_kernel<<<64, 256, 0, stream>>>(raw, ssum, ssq);
  bnparams_kernel<<<1, 128, 0, stream>>>(ssum, ssq, g1, be1, scale, shift, 64);
  bnrelu_kernel<64><<<bn_blocks, 256, 0, stream>>>(raw, scale, shift, Y2);

  // Layer 3: 64 -> 128, BN+ReLU+maxpool fused
  gemm_kernel<64, 128><<<gemm_blocks, 256, 0, stream>>>(Y2, w2, b2, raw, 64);
  stats_kernel<<<128, 256, 0, stream>>>(raw, ssum, ssq);
  bnparams_kernel<<<1, 128, 0, stream>>>(ssum, ssq, g2, be2, scale, shift, 128);
  maxpool_kernel<<<(128 * B_ * S_) / 256, 256, 0, stream>>>(raw, scale, shift, feat);
}
